// FusedMoE_70927089926455
// MI455X (gfx1250) — compile-verified, matrix operand order checked by reading, not run on hardware
//
#include <hip/hip_runtime.h>

// ---------------------------------------------------------------------------
// Fused MoE (top-2 of 8 experts), T=4096 H=2048 I=1408, bf16 WMMA compute.
// Double-buffered LDS pipeline, K-loop unrolled by 2 so buffer indices are
// compile-time constants (keeps all staging on ds_store/ds_load, no flat).
// ---------------------------------------------------------------------------

#define T_TOK 4096
#define H_DIM 2048
#define I_DIM 1408
#define E_NUM 8

typedef __bf16 bf16;
typedef __attribute__((ext_vector_type(4)))  __bf16 v4bf;
typedef __attribute__((ext_vector_type(8)))  __bf16 v8bf;
typedef __attribute__((ext_vector_type(16))) __bf16 v16bf;
typedef __attribute__((ext_vector_type(8)))  float  v8f;

union V16 { v16bf v; v8bf h8[2]; };

#define BM 128
#define BN 64
#define BK 32
#define LDSP (BK + 8)   // 40 bf16 = 80B row stride -> all 16B chunks aligned

// ---------------------------------------------------------------------------
__global__ void zero_counts_kernel(int* __restrict__ counts) {
  if (threadIdx.x < E_NUM) counts[threadIdx.x] = 0;
}

// fp32 -> bf16, 4 elements / thread
__global__ void cvt_hidden_kernel(const float* __restrict__ in,
                                  bf16* __restrict__ out) {
  const int i = blockIdx.x * blockDim.x + threadIdx.x;
  const float4 f = *(const float4*)(in + (size_t)i * 4);
  v4bf b = { (bf16)f.x, (bf16)f.y, (bf16)f.z, (bf16)f.w };
  *(v4bf*)(out + (size_t)i * 4) = b;
}

// Top-2 routing. softmax denominator cancels after top-k renormalization, so
// weights = softmax over the two top logits. Build per-expert gather lists.
__global__ void router_kernel(const float* __restrict__ logits,
                              int* __restrict__ counts,
                              int* __restrict__ lists,
                              float* __restrict__ w_of) {
  const int t = blockIdx.x * blockDim.x + threadIdx.x;
  if (t >= T_TOK) return;
  float l[E_NUM];
#pragma unroll
  for (int e = 0; e < E_NUM; ++e) l[e] = logits[t * E_NUM + e];
  int i0 = 0; float m0 = l[0];
#pragma unroll
  for (int e = 1; e < E_NUM; ++e) if (l[e] > m0) { m0 = l[e]; i0 = e; }
  int i1 = -1; float m1 = -3.4e38f;
#pragma unroll
  for (int e = 0; e < E_NUM; ++e)
    if (e != i0 && l[e] > m1) { m1 = l[e]; i1 = e; }
  const float e1  = __expf(m1 - m0);
  const float inv = __builtin_amdgcn_rcpf(1.0f + e1);
  w_of[2 * t + 0] = inv;
  w_of[2 * t + 1] = e1 * inv;
  const int p0 = atomicAdd(&counts[i0], 1);
  lists[i0 * T_TOK + p0] = 2 * t;        // entry id = 2*token + slot
  const int p1 = atomicAdd(&counts[i1], 1);
  lists[i1 * T_TOK + p1] = 2 * t + 1;
}

// ---------------------------------------------------------------------------
// GEMM1: gathered rows A (bf16 hidden) x w13^T; computes gate and up tiles
// sharing the A tile; epilogue silu(gate)*up -> act[entry][I] (bf16).
// ---------------------------------------------------------------------------
__global__ __launch_bounds__(256, 1) void gemm1_kernel(
    const bf16*  __restrict__ hs,
    const float* __restrict__ w13,
    const float* __restrict__ w13_scale,
    const int*   __restrict__ lists,
    const int*   __restrict__ counts,
    bf16*        __restrict__ act) {
  const int e    = blockIdx.z;
  const int cnt  = counts[e];
  const int row0 = blockIdx.y * BM;
  if (row0 >= cnt) return;
  const int n0      = blockIdx.x * BN;
  const float scale = w13_scale[e];

  __shared__ bf16 As[2][BM][LDSP];
  __shared__ bf16 Bgu[2][2][BN][LDSP];   // [buf][gate/up][row][k]
  __shared__ int  toks[BM];

  const int tid   = threadIdx.x;
  const int lane  = tid & 31;
  const int wid   = tid >> 5;
  const int wm    = wid & 3;          // 4 row groups of 32
  const int wn    = wid >> 2;         // 2 col groups of 32
  const int lrow  = lane & 15;
  const int lhalf = lane >> 4;

  if (tid < BM) {
    const int pos = row0 + tid;
    toks[tid] = (pos < cnt) ? lists[e * T_TOK + pos] : -1;
  }
  __syncthreads();

  // A staging: 128 rows x 32 k of bf16 -> 2x(16-elem chunk)/thread
  int a_row[2], a_kc[2];
  const bf16* a_ptr[2];
#pragma unroll
  for (int l = 0; l < 2; ++l) {
    const int idx = tid + l * 256;
    a_row[l] = idx >> 1;
    a_kc[l]  = (idx & 1) * 16;
    const int entry = toks[a_row[l]];
    const int tok   = (entry < 0) ? 0 : (entry >> 1);
    a_ptr[l] = hs + (size_t)tok * H_DIM + a_kc[l];
  }
  // B staging: gate + up, each 64 rows x 32 k fp32 -> 4x(float4)/thread
  int b_row[4], b_kc[4], b_mat[4];
  const float* b_ptr[4];
#pragma unroll
  for (int l = 0; l < 4; ++l) {
    const int idx = tid + l * 256;
    b_mat[l] = idx >> 9;              // 0 = gate, 1 = up
    const int rem = idx & 511;
    b_row[l] = rem >> 3;
    b_kc[l]  = (rem & 7) * 4;
    const int grow = b_mat[l] ? (I_DIM + n0 + b_row[l]) : (n0 + b_row[l]);
    b_ptr[l] = w13 + ((size_t)e * (2 * I_DIM) + grow) * H_DIM + b_kc[l];
  }

  v8bf   a_reg[2][2];
  float4 b_reg[4];
  v8f accg[2][2] = {};
  v8f accu[2][2] = {};

  // prologue: fetch tile 0 and stage into buffer 0
#pragma unroll
  for (int l = 0; l < 2; ++l) {
    const v8bf* src = (const v8bf*)a_ptr[l];
    a_reg[l][0] = src[0];
    a_reg[l][1] = src[1];
  }
#pragma unroll
  for (int l = 0; l < 4; ++l) b_reg[l] = *(const float4*)b_ptr[l];
#pragma unroll
  for (int l = 0; l < 2; ++l) {
    *(v8bf*)&As[0][a_row[l]][a_kc[l]]     = a_reg[l][0];
    *(v8bf*)&As[0][a_row[l]][a_kc[l] + 8] = a_reg[l][1];
  }
#pragma unroll
  for (int l = 0; l < 4; ++l) {
    const float4 f = b_reg[l];
    v4bf bv = { (bf16)(f.x * scale), (bf16)(f.y * scale),
                (bf16)(f.z * scale), (bf16)(f.w * scale) };
    *(v4bf*)&Bgu[0][b_mat[l]][b_row[l]][b_kc[l]] = bv;
  }
  __syncthreads();

  // one pipeline step; cur/nxt are always literal constants at the call site
  auto kstep = [&](const int cur, const int nxt, const int kt,
                   const bool more) __attribute__((always_inline)) {
    if (more) {
      const int k0 = (kt + 1) * BK;
#pragma unroll
      for (int l = 0; l < 2; ++l) {
        const v8bf* src = (const v8bf*)(a_ptr[l] + k0);
        a_reg[l][0] = src[0];
        a_reg[l][1] = src[1];
      }
#pragma unroll
      for (int l = 0; l < 4; ++l) b_reg[l] = *(const float4*)(b_ptr[l] + k0);
    }

    V16 a[2], bg[2], bu[2];
#pragma unroll
    for (int mi = 0; mi < 2; ++mi) {
      const int r = wm * 32 + mi * 16 + lrow;
      a[mi].h8[0] = *(const v8bf*)&As[cur][r][8 * lhalf];
      a[mi].h8[1] = *(const v8bf*)&As[cur][r][16 + 8 * lhalf];
    }
#pragma unroll
    for (int ni = 0; ni < 2; ++ni) {
      const int c = wn * 32 + ni * 16 + lrow;
      bg[ni].h8[0] = *(const v8bf*)&Bgu[cur][0][c][8 * lhalf];
      bg[ni].h8[1] = *(const v8bf*)&Bgu[cur][0][c][16 + 8 * lhalf];
      bu[ni].h8[0] = *(const v8bf*)&Bgu[cur][1][c][8 * lhalf];
      bu[ni].h8[1] = *(const v8bf*)&Bgu[cur][1][c][16 + 8 * lhalf];
    }
#pragma unroll
    for (int mi = 0; mi < 2; ++mi)
#pragma unroll
      for (int ni = 0; ni < 2; ++ni) {
        accg[mi][ni] = __builtin_amdgcn_wmma_f32_16x16x32_bf16(
            false, a[mi].v, false, bg[ni].v, (short)0, accg[mi][ni], false, false);
        accu[mi][ni] = __builtin_amdgcn_wmma_f32_16x16x32_bf16(
            false, a[mi].v, false, bu[ni].v, (short)0, accu[mi][ni], false, false);
      }

    if (more) {
#pragma unroll
      for (int l = 0; l < 2; ++l) {
        *(v8bf*)&As[nxt][a_row[l]][a_kc[l]]     = a_reg[l][0];
        *(v8bf*)&As[nxt][a_row[l]][a_kc[l] + 8] = a_reg[l][1];
      }
#pragma unroll
      for (int l = 0; l < 4; ++l) {
        const float4 f = b_reg[l];
        v4bf bv = { (bf16)(f.x * scale), (bf16)(f.y * scale),
                    (bf16)(f.z * scale), (bf16)(f.w * scale) };
        *(v4bf*)&Bgu[nxt][b_mat[l]][b_row[l]][b_kc[l]] = bv;
      }
    }
    __syncthreads();
  };

  const int KT = H_DIM / BK;          // 64, even
  for (int kt = 0; kt < KT; kt += 2) {
    kstep(0, 1, kt, true);
    kstep(1, 0, kt + 1, (kt + 2) < KT);
  }

  // epilogue: silu(gate) * up -> act  (rcp-based silu, no IEEE divide)
#pragma unroll
  for (int mi = 0; mi < 2; ++mi)
#pragma unroll
    for (int ni = 0; ni < 2; ++ni)
#pragma unroll
      for (int r = 0; r < 8; ++r) {
        const int rloc = wm * 32 + mi * 16 + r + 8 * lhalf;
        if (row0 + rloc < cnt) {
          const int entry = toks[rloc];
          const int col   = n0 + wn * 32 + ni * 16 + lrow;
          const float g = accg[mi][ni][r];
          const float u = accu[mi][ni][r];
          const float s = g * __builtin_amdgcn_rcpf(1.0f + __expf(-g)) * u;
          act[(size_t)entry * I_DIM + col] = (bf16)s;
        }
      }
}

// ---------------------------------------------------------------------------
// GEMM2: act (bf16, gathered) x w2^T -> y[entry][H] (fp32)
// ---------------------------------------------------------------------------
__global__ __launch_bounds__(256, 1) void gemm2_kernel(
    const bf16*  __restrict__ act,
    const float* __restrict__ w2,
    const float* __restrict__ w2_scale,
    const int*   __restrict__ lists,
    const int*   __restrict__ counts,
    float*       __restrict__ y) {
  const int e    = blockIdx.z;
  const int cnt  = counts[e];
  const int row0 = blockIdx.y * BM;
  if (row0 >= cnt) return;
  const int n0      = blockIdx.x * BN;   // output column (H) tile
  const float scale = w2_scale[e];

  __shared__ bf16 As[2][BM][LDSP];
  __shared__ bf16 Bs[2][BN][LDSP];
  __shared__ int  toks[BM];

  const int tid   = threadIdx.x;
  const int lane  = tid & 31;
  const int wid   = tid >> 5;
  const int wm    = wid & 3;
  const int wn    = wid >> 2;
  const int lrow  = lane & 15;
  const int lhalf = lane >> 4;

  if (tid < BM) {
    const int pos = row0 + tid;
    toks[tid] = (pos < cnt) ? lists[e * T_TOK + pos] : -1;
  }
  __syncthreads();

  int a_row[2], a_kc[2];
  const bf16* a_ptr[2];
#pragma unroll
  for (int l = 0; l < 2; ++l) {
    const int idx = tid + l * 256;
    a_row[l] = idx >> 1;
    a_kc[l]  = (idx & 1) * 16;
    const int entry = toks[a_row[l]];
    const int arow  = (entry < 0) ? 0 : entry;
    a_ptr[l] = act + (size_t)arow * I_DIM + a_kc[l];
  }
  int b_row[2], b_kc[2];
  const float* b_ptr[2];
#pragma unroll
  for (int l = 0; l < 2; ++l) {
    const int idx = tid + l * 256;
    b_row[l] = idx >> 3;
    b_kc[l]  = (idx & 7) * 4;
    b_ptr[l] = w2 + ((size_t)e * H_DIM + n0 + b_row[l]) * I_DIM + b_kc[l];
  }

  v8bf   a_reg[2][2];
  float4 b_reg[2];
  v8f acc[2][2] = {};

#pragma unroll
  for (int l = 0; l < 2; ++l) {
    const v8bf* src = (const v8bf*)a_ptr[l];
    a_reg[l][0] = src[0];
    a_reg[l][1] = src[1];
  }
#pragma unroll
  for (int l = 0; l < 2; ++l) b_reg[l] = *(const float4*)b_ptr[l];
#pragma unroll
  for (int l = 0; l < 2; ++l) {
    *(v8bf*)&As[0][a_row[l]][a_kc[l]]     = a_reg[l][0];
    *(v8bf*)&As[0][a_row[l]][a_kc[l] + 8] = a_reg[l][1];
  }
#pragma unroll
  for (int l = 0; l < 2; ++l) {
    const float4 f = b_reg[l];
    v4bf bv = { (bf16)(f.x * scale), (bf16)(f.y * scale),
                (bf16)(f.z * scale), (bf16)(f.w * scale) };
    *(v4bf*)&Bs[0][b_row[l]][b_kc[l]] = bv;
  }
  __syncthreads();

  auto kstep = [&](const int cur, const int nxt, const int kt,
                   const bool more) __attribute__((always_inline)) {
    if (more) {
      const int k0 = (kt + 1) * BK;
#pragma unroll
      for (int l = 0; l < 2; ++l) {
        const v8bf* src = (const v8bf*)(a_ptr[l] + k0);
        a_reg[l][0] = src[0];
        a_reg[l][1] = src[1];
      }
#pragma unroll
      for (int l = 0; l < 2; ++l) b_reg[l] = *(const float4*)(b_ptr[l] + k0);
    }

    V16 a[2], b[2];
#pragma unroll
    for (int mi = 0; mi < 2; ++mi) {
      const int r = wm * 32 + mi * 16 + lrow;
      a[mi].h8[0] = *(const v8bf*)&As[cur][r][8 * lhalf];
      a[mi].h8[1] = *(const v8bf*)&As[cur][r][16 + 8 * lhalf];
    }
#pragma unroll
    for (int ni = 0; ni < 2; ++ni) {
      const int c = wn * 32 + ni * 16 + lrow;
      b[ni].h8[0] = *(const v8bf*)&Bs[cur][c][8 * lhalf];
      b[ni].h8[1] = *(const v8bf*)&Bs[cur][c][16 + 8 * lhalf];
    }
#pragma unroll
    for (int mi = 0; mi < 2; ++mi)
#pragma unroll
      for (int ni = 0; ni < 2; ++ni)
        acc[mi][ni] = __builtin_amdgcn_wmma_f32_16x16x32_bf16(
            false, a[mi].v, false, b[ni].v, (short)0, acc[mi][ni], false, false);

    if (more) {
#pragma unroll
      for (int l = 0; l < 2; ++l) {
        *(v8bf*)&As[nxt][a_row[l]][a_kc[l]]     = a_reg[l][0];
        *(v8bf*)&As[nxt][a_row[l]][a_kc[l] + 8] = a_reg[l][1];
      }
#pragma unroll
      for (int l = 0; l < 2; ++l) {
        const float4 f = b_reg[l];
        v4bf bv = { (bf16)(f.x * scale), (bf16)(f.y * scale),
                    (bf16)(f.z * scale), (bf16)(f.w * scale) };
        *(v4bf*)&Bs[nxt][b_row[l]][b_kc[l]] = bv;
      }
    }
    __syncthreads();
  };

  const int KT = I_DIM / BK;          // 44, even
  for (int kt = 0; kt < KT; kt += 2) {
    kstep(0, 1, kt, true);
    kstep(1, 0, kt + 1, (kt + 2) < KT);
  }

#pragma unroll
  for (int mi = 0; mi < 2; ++mi)
#pragma unroll
    for (int ni = 0; ni < 2; ++ni)
#pragma unroll
      for (int r = 0; r < 8; ++r) {
        const int rloc = wm * 32 + mi * 16 + r + 8 * lhalf;
        if (row0 + rloc < cnt) {
          const int entry = toks[rloc];
          const int col   = n0 + wn * 32 + ni * 16 + lrow;
          y[(size_t)entry * H_DIM + col] = acc[mi][ni][r];
        }
      }
}

// out[t] = w0 * y[2t] + w1 * y[2t+1]   (4 floats / thread)
__global__ void combine_kernel(const float* __restrict__ y,
                               const float* __restrict__ w_of,
                               float* __restrict__ out) {
  const int i = blockIdx.x * blockDim.x + threadIdx.x;
  const int t = i / (H_DIM / 4);
  const int c = (i - t * (H_DIM / 4)) * 4;
  const float w0 = w_of[2 * t + 0];
  const float w1 = w_of[2 * t + 1];
  const float4 a = *(const float4*)(y + (size_t)(2 * t + 0) * H_DIM + c);
  const float4 b = *(const float4*)(y + (size_t)(2 * t + 1) * H_DIM + c);
  float4 o;
  o.x = w0 * a.x + w1 * b.x;
  o.y = w0 * a.y + w1 * b.y;
  o.z = w0 * a.z + w1 * b.z;
  o.w = w0 * a.w + w1 * b.w;
  *(float4*)(out + (size_t)t * H_DIM + c) = o;
}

// ---------------------------------------------------------------------------
extern "C" void kernel_launch(void* const* d_in, const int* in_sizes, int n_in,
                              void* d_out, int out_size, void* d_ws, size_t ws_size,
                              hipStream_t stream) {
  const float* hs   = (const float*)d_in[0];
  const float* rl   = (const float*)d_in[1];
  const float* w13  = (const float*)d_in[2];
  const float* w13s = (const float*)d_in[3];
  const float* w2   = (const float*)d_in[4];
  const float* w2s  = (const float*)d_in[5];
  float* out = (float*)d_out;

  // workspace layout (256B aligned)
  char* ws = (char*)d_ws;
  auto take = [&](size_t bytes) -> char* {
    char* p = ws;
    ws += (bytes + 255) & ~(size_t)255;
    return p;
  };
  bf16*  hsb   = (bf16*)take((size_t)T_TOK * H_DIM * sizeof(bf16));          // 16 MB
  bf16*  actb  = (bf16*)take((size_t)2 * T_TOK * I_DIM * sizeof(bf16));      // 22 MB
  float* yb    = (float*)take((size_t)2 * T_TOK * H_DIM * sizeof(float));    // 64 MB
  int*   lists = (int*)take((size_t)E_NUM * T_TOK * sizeof(int));
  float* w_of  = (float*)take((size_t)T_TOK * 2 * sizeof(float));
  int*   cnts  = (int*)take((size_t)E_NUM * sizeof(int));
  (void)ws_size; (void)in_sizes; (void)n_in; (void)out_size;

  zero_counts_kernel<<<1, 64, 0, stream>>>(cnts);
  cvt_hidden_kernel<<<(T_TOK * H_DIM / 4) / 256, 256, 0, stream>>>(hs, hsb);
  router_kernel<<<T_TOK / 256, 256, 0, stream>>>(rl, cnts, lists, w_of);
  gemm1_kernel<<<dim3(I_DIM / BN, T_TOK / BM, E_NUM), 256, 0, stream>>>(
      hsb, w13, w13s, lists, cnts, actb);
  gemm2_kernel<<<dim3(H_DIM / BN, T_TOK / BM, E_NUM), 256, 0, stream>>>(
      actb, w2, w2s, lists, cnts, yb);
  combine_kernel<<<(T_TOK * H_DIM / 4) / 256, 256, 0, stream>>>(yb, w_of, out);
}
